// Mamba2Block_80805514707243
// MI455X (gfx1250) — matrix operand order repaired and checked by reading, not compiled, not run on hardware
//
#include <hip/hip_runtime.h>
#include <hip/hip_bf16.h>
#include <cmath>

#define B_      2
#define L_      4096
#define DMODEL  1024
#define DINNER  2048
#define NHEADS  32
#define HEADDIM 64
#define DSTATE  128
#define CONVDIM 2304
#define NPROJ   4384
#define ZLD     4480     // NPROJ padded to a multiple of 128 for the GEMM
#define NCHUNK  16
#define CHUNK   256
#define ROWS    (B_ * L_)

typedef _Float16 v16h __attribute__((ext_vector_type(16)));
typedef _Float16 v8h  __attribute__((ext_vector_type(8)));
typedef float    v8f  __attribute__((ext_vector_type(8)));

// ---- WMMA fragment helpers (CDNA5 16x16x32 f16 layouts, ISA 7.12.2/7.12.4) ----
// A fragment from a row-major [16 x K] tile (stride in halves, multiple of 8):
// lane m=lane&15, kb=8*(lane>>4): frag[0..7]=row[kb..kb+7], frag[8..15]=row[kb+16..kb+23]
__device__ __forceinline__ v16h frag_a(const _Float16* p, int stride) {
  const int lane = threadIdx.x & 31;
  const _Float16* r = p + (lane & 15) * stride + (lane >> 4) * 8;
  v8h lo = *(const v8h*)r;
  v8h hi = *(const v8h*)(r + 16);
  v16h f;
#pragma unroll
  for (int i = 0; i < 8; ++i) { f[i] = lo[i]; f[i + 8] = hi[i]; }
  return f;
}
// B fragment (K x 16 matrix) from its transpose stored row-major [16(N) x K]:
// lane n=lane&15, kb=8*(lane>>4): frag[0..7]=B[kb..kb+7][n], frag[8..15]=B[kb+16..kb+23][n]
__device__ __forceinline__ v16h frag_b(const _Float16* p, int stride) {
  const int lane = threadIdx.x & 31;
  const _Float16* r = p + (lane & 15) * stride + (lane >> 4) * 8;
  v8h lo = *(const v8h*)r;
  v8h hi = *(const v8h*)(r + 16);
  v16h f;
#pragma unroll
  for (int i = 0; i < 8; ++i) { f[i] = lo[i]; f[i + 8] = hi[i]; }
  return f;
}
__device__ __forceinline__ v8f wmma32(v16h a, v16h b, v8f c) {
  return __builtin_amdgcn_wmma_f32_16x16x32_f16(false, a, false, b, (short)0, c,
                                                false, false);
}

// ---- K1: cast f32 [K][N] weight -> f16 [Npad][K] transposed, zero-padded rows
__global__ __launch_bounds__(256) void k_cast_t(const float* __restrict__ src,
                                                _Float16* __restrict__ dst,
                                                int K, int N, int Npad) {
  int idx = blockIdx.x * 256 + threadIdx.x;
  if (idx >= K * Npad) return;
  int n = idx / K, k = idx - n * K;
  dst[idx] = (n < N) ? (_Float16)src[(size_t)k * N + n] : (_Float16)0.f;
}

// ---- K2: input RMSNorm -> f16 activations
__global__ __launch_bounds__(256) void k_rmsnorm_in(const float* __restrict__ x,
                                                    const float* __restrict__ w,
                                                    _Float16* __restrict__ o) {
  __shared__ float red[256];
  int row = blockIdx.x, tid = threadIdx.x;
  const float* xr = x + (size_t)row * DMODEL;
  float v[4]; float ss = 0.f;
#pragma unroll
  for (int j = 0; j < 4; ++j) { v[j] = xr[j * 256 + tid]; ss += v[j] * v[j]; }
  red[tid] = ss; __syncthreads();
  for (int s = 128; s > 0; s >>= 1) { if (tid < s) red[tid] += red[tid + s]; __syncthreads(); }
  float sc = rsqrtf(red[0] / (float)DMODEL + 1e-5f);
#pragma unroll
  for (int j = 0; j < 4; ++j) {
    int c = j * 256 + tid;
    o[(size_t)row * DMODEL + c] = (_Float16)(v[j] * sc * w[c]);
  }
}

// ---- WMMA GEMM: C[M,Npad] = A[M,K](f16) * Bt[Npad,K](f16)^T (+residual)
// block tile 128x128, 8 waves (2 in M x 4 in N), wave tile 64x32, K-step 32
__global__ __launch_bounds__(256) void k_gemm(const _Float16* __restrict__ A,
                                              const _Float16* __restrict__ Bt,
                                              float* __restrict__ C,
                                              const float* __restrict__ resid,
                                              int K, int ldc) {
  __shared__ __align__(16) _Float16 As[128 * 40];
  __shared__ __align__(16) _Float16 Bs[128 * 40];
  const int tid = threadIdx.x;
  const int w = tid >> 5, lane = tid & 31;
  const int mw = w & 1, nw = w >> 1;
  const int m0 = blockIdx.y * 128, n0 = blockIdx.x * 128;
  v8f acc[4][2] = {};
  for (int k0 = 0; k0 < K; k0 += 32) {
    __syncthreads();
#pragma unroll
    for (int i = 0; i < 2; ++i) {
      int t = tid + i * 256;
      int row = t >> 2, col = (t & 3) * 8;
      *(v8h*)&As[row * 40 + col] = *(const v8h*)&A[(size_t)(m0 + row) * K + k0 + col];
      *(v8h*)&Bs[row * 40 + col] = *(const v8h*)&Bt[(size_t)(n0 + row) * K + k0 + col];
      if (k0 + 32 < K) {   // CDNA5 global_prefetch of next K tile
        __builtin_prefetch(&A[(size_t)(m0 + row) * K + k0 + 32 + col], 0, 0);
        __builtin_prefetch(&Bt[(size_t)(n0 + row) * K + k0 + 32 + col], 0, 0);
      }
    }
    __syncthreads();
    v16h af[4], bf[2];
#pragma unroll
    for (int i = 0; i < 4; ++i) af[i] = frag_a(&As[(mw * 64 + i * 16) * 40], 40);
#pragma unroll
    for (int j = 0; j < 2; ++j) bf[j] = frag_b(&Bs[(nw * 32 + j * 16) * 40], 40);
#pragma unroll
    for (int i = 0; i < 4; ++i)
#pragma unroll
      for (int j = 0; j < 2; ++j)
        acc[i][j] = wmma32(af[i], bf[j], acc[i][j]);
  }
  const int hi = lane >> 4, nn = lane & 15;
#pragma unroll
  for (int i = 0; i < 4; ++i)
#pragma unroll
    for (int j = 0; j < 2; ++j)
#pragma unroll
      for (int v = 0; v < 8; ++v) {
        int m = m0 + mw * 64 + i * 16 + hi * 8 + v;
        int n = n0 + nw * 32 + j * 16 + nn;
        float val = acc[i][j][v];
        if (resid) val += resid[(size_t)m * ldc + n];
        C[(size_t)m * ldc + n] = val;
      }
}

// ---- K4: causal depthwise conv (width 4) + bias + SiLU -> xBC f16
__global__ __launch_bounds__(256) void k_conv(const float* __restrict__ zx,
                                              const float* __restrict__ cw,
                                              const float* __restrict__ cb,
                                              _Float16* __restrict__ xBC) {
  int idx = blockIdx.x * 256 + threadIdx.x;
  if (idx >= ROWS * CONVDIM) return;
  int ch = idx % CONVDIM;
  int row = idx / CONVDIM;       // b*L + l
  int l = row & (L_ - 1);
  float acc = cb[ch];
#pragma unroll
  for (int j = 0; j < 4; ++j) {
    int ls = l + j - 3;
    if (ls >= 0) acc += cw[ch * 4 + j] * zx[(size_t)(row + j - 3) * ZLD + DINNER + ch];
  }
  float s = acc / (1.f + __expf(-acc));
  xBC[(size_t)row * CONVDIM + ch] = (_Float16)s;
}

// ---- K5: dt softplus and dt*A
__global__ __launch_bounds__(256) void k_dt(const float* __restrict__ zx,
                                            const float* __restrict__ dtb,
                                            const float* __restrict__ Alog,
                                            float* __restrict__ dtv,
                                            float* __restrict__ dtA) {
  int idx = blockIdx.x * 256 + threadIdx.x;
  if (idx >= ROWS * NHEADS) return;
  int h = idx & 31, row = idx >> 5;
  float x = zx[(size_t)row * ZLD + (DINNER + CONVDIM) + h] + dtb[h];
  float sp = (x > 20.f) ? x : log1pf(__expf(x));
  dtv[idx] = sp;
  dtA[idx] = sp * (-__expf(Alog[h]));
}

// ---- K6: SSD pass1 per (b,c,h): cumsum, Y_diag (C.B^T masked * xdt), chunk states
__global__ __launch_bounds__(256) void k_ssd1(const _Float16* __restrict__ xBC,
                                              const float* __restrict__ dtv_g,
                                              const float* __restrict__ dtA_g,
                                              float* __restrict__ acum_g,
                                              float* __restrict__ statesT,
                                              float* __restrict__ y_g) {
  __shared__ float acum[CHUNK];
  __shared__ float dts[CHUNK];
  __shared__ __align__(16) _Float16 xdtT[64 * 264];      // [p][l], stride 264
  __shared__ __align__(16) _Float16 stg[8 * 16 * 40];    // per-wave 16x32 staging
  const int h = blockIdx.x, c = blockIdx.y, b = blockIdx.z;
  const int tid = threadIdx.x, w = tid >> 5, lane = tid & 31;
  const size_t row0 = (size_t)b * L_ + (size_t)c * CHUNK;
  dts[tid]  = dtv_g[(row0 + tid) * NHEADS + h];
  acum[tid] = dtA_g[(row0 + tid) * NHEADS + h];
  __syncthreads();
  // parallel inclusive scan (Hillis-Steele, 8 steps)
#pragma unroll
  for (int off = 1; off < CHUNK; off <<= 1) {
    float t = (tid >= off) ? acum[tid - off] : 0.f;
    __syncthreads();
    acum[tid] += t;
    __syncthreads();
  }
  const size_t abase = ((size_t)(b * NCHUNK + c) * NHEADS + h) * CHUNK;
  acum_g[abase + tid] = acum[tid];
  for (int i = tid; i < 64 * CHUNK; i += 256) {
    int p = i & 63, l = i >> 6;
    float xv = (float)xBC[(row0 + l) * CONVDIM + h * HEADDIM + p];
    xdtT[p * 264 + l] = (_Float16)(xv * dts[l]);
  }
  __syncthreads();

  const _Float16* Bg = xBC + row0 * CONVDIM + DINNER;
  const _Float16* Cg = Bg + DSTATE;
  _Float16* wstg = &stg[w * 16 * 40];
  const int lb = w * 32;                 // wave owns l rows [lb, lb+32)
  const int M0 = (lane >> 4) * 8, nn = lane & 15;

  // ---- Y_diag (C fragments hoisted: loaded once per 16-row band) ----
#pragma unroll
  for (int ls = 0; ls < 2; ++ls) {
    const int lr = lb + ls * 16;
    v16h ca[4];
#pragma unroll
    for (int i = 0; i < 4; ++i)
      ca[i] = frag_a(Cg + (size_t)lr * CONVDIM + i * 32, CONVDIM);
    v8f ydl[4] = {};
    for (int s0 = 0; s0 <= lb; s0 += 32) {
#pragma unroll
      for (int ssb = 0; ssb < 2; ++ssb) {
        v8f g = {};
#pragma unroll
        for (int i = 0; i < 4; ++i) {
          v16h bb = frag_b(Bg + (size_t)(s0 + ssb * 16) * CONVDIM + i * 32, CONVDIM);
          g = wmma32(ca[i], bb, g);
        }
        int sg = s0 + ssb * 16 + nn;
#pragma unroll
        for (int v = 0; v < 8; ++v) {
          int lg = lr + M0 + v;
          float val = (sg <= lg) ? g[v] * __expf(acum[lg] - acum[sg]) : 0.f;
          wstg[(M0 + v) * 40 + ssb * 16 + nn] = (_Float16)val;
        }
      }
      v16h ga = frag_a(wstg, 40);
#pragma unroll
      for (int ps = 0; ps < 4; ++ps) {
        v16h xb = frag_b(&xdtT[(ps * 16) * 264 + s0], 264);
        ydl[ps] = wmma32(ga, xb, ydl[ps]);
      }
    }
#pragma unroll
    for (int ps = 0; ps < 4; ++ps)
#pragma unroll
      for (int v = 0; v < 8; ++v) {
        int lg = lr + M0 + v;
        int p = ps * 16 + nn;
        y_g[(row0 + lg) * DINNER + h * HEADDIM + p] = ydl[ps][v];
      }
  }

  // ---- chunk states: statesT[n,p] = sum_l (B[l,n]*decay_l) * xdt[l,p] ----
  v8f st[4] = {};
  float alast = acum[CHUNK - 1];
  for (int l0 = 0; l0 < CHUNK; l0 += 32) {
    for (int i = lane; i < 512; i += 32) {      // stage decay-folded B^T 16x32
      int n = i & 15, li = i >> 4;
      float bv = (float)Bg[(size_t)(l0 + li) * CONVDIM + w * 16 + n];
      wstg[n * 40 + li] = (_Float16)(bv * __expf(alast - acum[l0 + li]));
    }
    v16h ab = frag_a(wstg, 40);
#pragma unroll
    for (int ps = 0; ps < 4; ++ps) {
      v16h xb = frag_b(&xdtT[(ps * 16) * 264 + l0], 264);
      st[ps] = wmma32(ab, xb, st[ps]);
    }
  }
  const size_t sbase = ((size_t)(b * NCHUNK + c) * NHEADS + h) * (DSTATE * HEADDIM);
#pragma unroll
  for (int ps = 0; ps < 4; ++ps)
#pragma unroll
    for (int v = 0; v < 8; ++v) {
      int n = w * 16 + M0 + v;
      int p = ps * 16 + nn;
      statesT[sbase + n * HEADDIM + p] = st[ps][v];
    }
}

// ---- K7: SSD pass2 per (b,h): sequential chunk scan, emit prev states [p][n] f16
__global__ __launch_bounds__(256) void k_ssd2(const float* __restrict__ statesT,
                                              const float* __restrict__ acum_g,
                                              _Float16* __restrict__ prevT) {
  const int h = blockIdx.x, b = blockIdx.y;
  const int tid = threadIdx.x;
  float run[32];
#pragma unroll
  for (int j = 0; j < 32; ++j) run[j] = 0.f;
  for (int c = 0; c < NCHUNK; ++c) {
    const size_t base = (size_t)(b * NCHUNK + c) * NHEADS + h;
#pragma unroll
    for (int j = 0; j < 32; ++j) {
      int idx = j * 256 + tid;
      int n = idx >> 6, p = idx & 63;
      prevT[base * (HEADDIM * DSTATE) + p * DSTATE + n] = (_Float16)run[j];
    }
    float sc = __expf(acum_g[base * CHUNK + (CHUNK - 1)]);
#pragma unroll
    for (int j = 0; j < 32; ++j)
      run[j] = run[j] * sc + statesT[base * (DSTATE * HEADDIM) + j * 256 + tid];
  }
}

// ---- K8: SSD pass3: Y_off = C * prev^T scaled by exp(acum); + D_skip * xs
__global__ __launch_bounds__(256) void k_ssd3(const _Float16* __restrict__ xBC,
                                              const _Float16* __restrict__ prevT,
                                              const float* __restrict__ acum_g,
                                              const float* __restrict__ Dskip,
                                              float* __restrict__ y_g) {
  __shared__ float acum[CHUNK];
  const int h = blockIdx.x, c = blockIdx.y, b = blockIdx.z;
  const int tid = threadIdx.x, w = tid >> 5, lane = tid & 31;
  const size_t row0 = (size_t)b * L_ + (size_t)c * CHUNK;
  const size_t base = (size_t)(b * NCHUNK + c) * NHEADS + h;
  acum[tid] = acum_g[base * CHUNK + tid];
  __syncthreads();
  const _Float16* Cg = xBC + row0 * CONVDIM + DINNER + DSTATE;
  const _Float16* Pg = prevT + base * (HEADDIM * DSTATE);
  const int lb = w * 32;
  v8f yo[2][4] = {};
#pragma unroll
  for (int n0 = 0; n0 < DSTATE; n0 += 32) {
    v16h bp[4];
#pragma unroll
    for (int ps = 0; ps < 4; ++ps)
      bp[ps] = frag_b(Pg + (ps * 16) * DSTATE + n0, DSTATE);
#pragma unroll
    for (int ls = 0; ls < 2; ++ls) {
      v16h ca = frag_a(Cg + (size_t)(lb + ls * 16) * CONVDIM + n0, CONVDIM);
#pragma unroll
      for (int ps = 0; ps < 4; ++ps)
        yo[ls][ps] = wmma32(ca, bp[ps], yo[ls][ps]);
    }
  }
  float Dv = Dskip[h];
  const int M0 = (lane >> 4) * 8, nn = lane & 15;
#pragma unroll
  for (int ls = 0; ls < 2; ++ls)
#pragma unroll
    for (int ps = 0; ps < 4; ++ps)
#pragma unroll
      for (int v = 0; v < 8; ++v) {
        int lg = lb + ls * 16 + M0 + v;
        int p = ps * 16 + nn;
        size_t yi = (row0 + lg) * DINNER + h * HEADDIM + p;
        float xs = (float)xBC[(row0 + lg) * CONVDIM + h * HEADDIM + p];
        y_g[yi] += yo[ls][ps][v] * __expf(acum[lg]) + xs * Dv;
      }
}

// ---- K9: gate (y * silu(z)) + RMSNorm over D_INNER -> f16
__global__ __launch_bounds__(256) void k_gate(const float* __restrict__ y_g,
                                              const float* __restrict__ zx,
                                              const float* __restrict__ gw,
                                              _Float16* __restrict__ og) {
  __shared__ float red[256];
  int row = blockIdx.x, tid = threadIdx.x;
  float g[8]; float ss = 0.f;
#pragma unroll
  for (int j = 0; j < 8; ++j) {
    int col = j * 256 + tid;
    float z = zx[(size_t)row * ZLD + col];
    float sz = z / (1.f + __expf(-z));
    g[j] = y_g[(size_t)row * DINNER + col] * sz;
    ss += g[j] * g[j];
  }
  red[tid] = ss; __syncthreads();
  for (int s = 128; s > 0; s >>= 1) { if (tid < s) red[tid] += red[tid + s]; __syncthreads(); }
  float sc = rsqrtf(red[0] / (float)DINNER + 1e-5f);
#pragma unroll
  for (int j = 0; j < 8; ++j) {
    int col = j * 256 + tid;
    og[(size_t)row * DINNER + col] = (_Float16)(g[j] * sc * gw[col]);
  }
}

extern "C" void kernel_launch(void* const* d_in, const int* in_sizes, int n_in,
                              void* d_out, int out_size, void* d_ws, size_t ws_size,
                              hipStream_t stream) {
  (void)in_sizes; (void)n_in; (void)out_size; (void)ws_size;
  const float* hid  = (const float*)d_in[0];
  const float* nw   = (const float*)d_in[1];
  const float* ipw  = (const float*)d_in[2];
  const float* cw   = (const float*)d_in[3];
  const float* cb   = (const float*)d_in[4];
  const float* dtb  = (const float*)d_in[5];
  const float* Alog = (const float*)d_in[6];
  const float* Dsk  = (const float*)d_in[7];
  const float* gw   = (const float*)d_in[8];
  const float* opw  = (const float*)d_in[9];
  float* out = (float*)d_out;

  char* wp = (char*)d_ws;
  auto alloc = [&](size_t bytes) {
    char* p = wp; wp += (bytes + 255) & ~(size_t)255; return p;
  };
  _Float16* Wt1   = (_Float16*)alloc((size_t)ZLD * DMODEL * 2);
  _Float16* Wt2   = (_Float16*)alloc((size_t)DMODEL * DINNER * 2);
  _Float16* xn    = (_Float16*)alloc((size_t)ROWS * DMODEL * 2);
  float*    zx    = (float*)   alloc((size_t)ROWS * ZLD * 4);
  _Float16* xBC   = (_Float16*)alloc((size_t)ROWS * CONVDIM * 2);
  float*    dtv   = (float*)   alloc((size_t)ROWS * NHEADS * 4);
  float*    dtA   = (float*)   alloc((size_t)ROWS * NHEADS * 4);
  float*    acum  = (float*)   alloc((size_t)ROWS * NHEADS * 4);
  float*    stT   = (float*)   alloc((size_t)B_ * NCHUNK * NHEADS * DSTATE * HEADDIM * 4);
  _Float16* prevT = (_Float16*)alloc((size_t)B_ * NCHUNK * NHEADS * DSTATE * HEADDIM * 2);
  float*    y     = (float*)   alloc((size_t)ROWS * DINNER * 4);
  _Float16* og    = (_Float16*)alloc((size_t)ROWS * DINNER * 2);

  k_cast_t<<<(ZLD * DMODEL + 255) / 256, 256, 0, stream>>>(ipw, Wt1, DMODEL, NPROJ, ZLD);
  k_cast_t<<<(DMODEL * DINNER + 255) / 256, 256, 0, stream>>>(opw, Wt2, DINNER, DMODEL, DMODEL);
  k_rmsnorm_in<<<ROWS, 256, 0, stream>>>(hid, nw, xn);
  k_gemm<<<dim3(ZLD / 128, ROWS / 128), 256, 0, stream>>>(xn, Wt1, zx, nullptr,
                                                          DMODEL, ZLD);
  k_conv<<<(ROWS * CONVDIM + 255) / 256, 256, 0, stream>>>(zx, cw, cb, xBC);
  k_dt<<<(ROWS * NHEADS + 255) / 256, 256, 0, stream>>>(zx, dtb, Alog, dtv, dtA);
  k_ssd1<<<dim3(NHEADS, NCHUNK, B_), 256, 0, stream>>>(xBC, dtv, dtA, acum, stT, y);
  k_ssd2<<<dim3(NHEADS, B_), 256, 0, stream>>>(stT, acum, prevT);
  k_ssd3<<<dim3(NHEADS, NCHUNK, B_), 256, 0, stream>>>(xBC, prevT, acum, Dsk, y);
  k_gate<<<ROWS, 256, 0, stream>>>(y, zx, gw, og);
  k_gemm<<<dim3(DMODEL / 128, ROWS / 128), 256, 0, stream>>>(og, Wt2, out, hid,
                                                             DINNER, DMODEL);
}
